// _MambaStack_60189671686602
// MI455X (gfx1250) — compile-verified, matrix operand order checked by reading, not run on hardware
//
#include <hip/hip_runtime.h>
#include <hip/hip_bf16.h>
#include <math.h>

// ---- Problem constants (match reference) ----
#define DEPTH   2
#define D_MODEL 768
#define D_STATE 64
#define D_CONV  4
#define D_INNER 1536
#define DT_RANK 48
#define BATCH   4
#define SEQ     2048
#define MROWS   (BATCH * SEQ)            // 8192
#define XDBL_W  (DT_RANK + 2 * D_STATE)  // 176

// ---- WMMA vector types (CDNA5 wave32) ----
typedef __attribute__((ext_vector_type(16))) __bf16 v16bf;
typedef __attribute__((ext_vector_type(8)))  float  v8f;

__device__ __forceinline__ __bf16 f2bf(float f) {
    unsigned u = __builtin_bit_cast(unsigned, f);
    u += 0x7fffu + ((u >> 16) & 1u);          // round-to-nearest-even
    unsigned short h = (unsigned short)(u >> 16);
    return __builtin_bit_cast(__bf16, h);
}

// ---- CDNA5 async global->LDS copy (ASYNCcnt-tracked, no VGPR round trip) ----
__device__ __forceinline__ void async_b128(void* lds, const void* gaddr) {
    asm volatile("global_load_async_to_lds_b128 %0, %1, off"
                 :: "v"((unsigned)(unsigned long long)lds), "v"(gaddr)
                 : "memory");
}
__device__ __forceinline__ void wait_async_le4() {
    asm volatile("s_wait_asynccnt 0x4" ::: "memory");
}
__device__ __forceinline__ void wait_async_0() {
    asm volatile("s_wait_asynccnt 0x0" ::: "memory");
}

// =====================================================================
// WMMA GEMM:  C[M,N] = A[M,K(lda)] * W[N,K(ldw)]^T   (bf16 in, fp32 acc)
// Block 256 thr = 8 waves, tile 128x128, K-slab 32. Wave grid 2(M)x4(N),
// each wave owns 64x32 = 4x2 wmma tiles. ALIGNED path: no bounds checks,
// async double-buffered staging. Generic path: 16B-chunk guards, single buf.
// =====================================================================
#define BM   128
#define BN   128
#define BKK  32
#define LSTR 40   // padded bf16 row stride (32 + 8) -> 80B rows

template<bool ALIGNED>
__device__ __forceinline__ void stage_tile(
    __bf16* __restrict__ sdst, const __bf16* __restrict__ G,
    int ld, int base, int rows_bound, int k0, int K, int tid)
{
    #pragma unroll
    for (int i = 0; i < 2; ++i) {
        int chunk = tid + i * 256;          // 0..511 : 16B chunks of the tile
        int r  = chunk >> 2;                // 0..127
        int c8 = (chunk & 3) << 3;          // 0,8,16,24 (bf16 elems)
        __bf16* l = sdst + r * LSTR + c8;   // 16B-aligned LDS dest
        if (ALIGNED) {
            async_b128(l, G + (size_t)(base + r) * ld + (k0 + c8));
        } else {
            int gr = base + r, gk = k0 + c8;
            if (gr < rows_bound && gk + 7 < K) {
                async_b128(l, G + (size_t)gr * ld + gk);
            } else {
                *(uint4*)l = make_uint4(0u, 0u, 0u, 0u);   // OOB -> zeros
            }
        }
    }
}

template<bool ALIGNED>
__global__ __launch_bounds__(256) void wmma_gemm_kernel(
    const __bf16* __restrict__ A, int lda,
    const __bf16* __restrict__ W, int ldw,
    float* __restrict__ C, __bf16* __restrict__ C16, int ldc,
    int M, int N, int K,
    const float* __restrict__ bias, int softplus_ep)
{
    __shared__ __bf16 sA[2][BM * LSTR];
    __shared__ __bf16 sW[2][BN * LSTR];

    const int tid   = threadIdx.x;
    const int lane  = tid & 31;
    const int wave  = tid >> 5;
    const int waveM = wave >> 2;          // 0..1  -> 64-row stripe
    const int waveN = wave & 3;           // 0..3  -> 32-col stripe
    const int lm    = lane & 15;          // row/col within 16x16 tile
    const int kb    = (lane >> 4) << 3;   // 0/8 : K-half select (A/B frag)
    const int rsel  = (lane >> 4) << 3;   // 0/8 : M offset (C/D layout)

    const int blockM = blockIdx.y * BM;
    const int blockN = blockIdx.x * BN;

    v8f acc[4][2] = {};
    const int nslab = (K + BKK - 1) / BKK;

    if (ALIGNED) {   // prologue: stage slab 0 into buffer 0
        stage_tile<true>(sA[0], A, lda, blockM, M, 0, K, tid);
        stage_tile<true>(sW[0], W, ldw, blockN, N, 0, K, tid);
    }

    for (int s = 0; s < nslab; ++s) {
        int cb;
        if (ALIGNED) {
            cb = s & 1;
            if (s + 1 < nslab) {
                __syncthreads();   // everyone done reading buf[cb^1] (slab s-1)
                stage_tile<true>(sA[cb ^ 1], A, lda, blockM, M, (s + 1) * BKK, K, tid);
                stage_tile<true>(sW[cb ^ 1], W, ldw, blockN, N, (s + 1) * BKK, K, tid);
                wait_async_le4();  // 4 newer loads outstanding; slab s is in LDS
            } else {
                wait_async_0();
            }
            __syncthreads();       // all waves' slab-s data visible
        } else {
            cb = 0;
            __syncthreads();       // prior compute done before overwrite
            stage_tile<false>(sA[0], A, lda, blockM, M, s * BKK, K, tid);
            stage_tile<false>(sW[0], W, ldw, blockN, N, s * BKK, K, tid);
            wait_async_0();
            __syncthreads();
        }

        // ---- A fragments (16-bit 16x32 layout, ISA 7.12.2) ----
        v16bf afrag[4];
        #pragma unroll
        for (int mt = 0; mt < 4; ++mt) {
            const __bf16* p = &sA[cb][(waveM * 64 + mt * 16 + lm) * LSTR];
            #pragma unroll
            for (int e = 0; e < 8; ++e) afrag[mt][e] = p[kb + e];
            #pragma unroll
            for (int e = 0; e < 8; ++e) afrag[mt][8 + e] = p[16 + kb + e];
        }
        #pragma unroll
        for (int nt = 0; nt < 2; ++nt) {
            v16bf bfrag;
            const __bf16* p = &sW[cb][(waveN * 32 + nt * 16 + lm) * LSTR];
            #pragma unroll
            for (int e = 0; e < 8; ++e) bfrag[e] = p[kb + e];
            #pragma unroll
            for (int e = 0; e < 8; ++e) bfrag[8 + e] = p[16 + kb + e];
            #pragma unroll
            for (int mt = 0; mt < 4; ++mt) {
                acc[mt][nt] = __builtin_amdgcn_wmma_f32_16x16x32_bf16(
                    false, afrag[mt], false, bfrag,
                    (short)0, acc[mt][nt], false, false);
            }
        }
    }

    // ---- epilogue + store (C/D layout: vgpr j -> M = j + 8*(lane>>4)) ----
    #pragma unroll
    for (int mt = 0; mt < 4; ++mt) {
        #pragma unroll
        for (int nt = 0; nt < 2; ++nt) {
            int gn = blockN + waveN * 32 + nt * 16 + lm;
            #pragma unroll
            for (int j = 0; j < 8; ++j) {
                int gm = blockM + waveM * 64 + mt * 16 + rsel + j;
                if ((ALIGNED || (gm < M && gn < N))) {
                    float v = acc[mt][nt][j];
                    if (softplus_ep) {
                        v += bias[gn];
                        v = (v > 20.f) ? v : __logf(1.f + __expf(v));
                    }
                    size_t o = (size_t)gm * ldc + gn;
                    if (C)   C[o]   = v;
                    if (C16) C16[o] = f2bf(v);
                }
            }
        }
    }
}

// =====================================================================
// f32 -> bf16 cast (weights / initial input)
// =====================================================================
__global__ __launch_bounds__(256) void cast_bf16_kernel(
    const float* __restrict__ in, __bf16* __restrict__ out, int n)
{
    int i = blockIdx.x * 256 + threadIdx.x;
    if (i < n) out[i] = f2bf(in[i]);
}

// =====================================================================
// Depthwise causal conv (K=4) + SiLU. xz = [B*L, 2*D_INNER]; writes f32 + bf16
// =====================================================================
__global__ __launch_bounds__(256) void conv_silu_kernel(
    const float* __restrict__ xz, const float* __restrict__ cw,
    const float* __restrict__ cb, float* __restrict__ out,
    __bf16* __restrict__ out16)
{
    int idx = blockIdx.x * 256 + threadIdx.x;
    if (idx >= MROWS * D_INNER) return;
    int d   = idx % D_INNER;
    int row = idx / D_INNER;          // b*SEQ + t
    int t   = row % SEQ;
    float acc = cb[d];
    if (t >= D_CONV - 1) {            // uniform fast path for most rows
        const float* p = xz + (size_t)(row - (D_CONV - 1)) * (2 * D_INNER) + d;
        #pragma unroll
        for (int k = 0; k < D_CONV; ++k)
            acc += p[(size_t)k * (2 * D_INNER)] * cw[d * D_CONV + k];
    } else {
        #pragma unroll
        for (int k = 0; k < D_CONV; ++k) {
            int tt = t - (D_CONV - 1) + k;
            if (tt >= 0)
                acc += xz[(size_t)(row - (D_CONV - 1) + k) * (2 * D_INNER) + d]
                       * cw[d * D_CONV + k];
        }
    }
    float r = acc / (1.f + __expf(-acc));   // SiLU
    out[idx]   = r;
    out16[idx] = f2bf(r);
}

// =====================================================================
// Selective scan: thread per (b,d); 64 states + A in VGPRs; B/C chunks of
// 32 timesteps staged LDS via async global->LDS; inner reads are broadcasts.
// =====================================================================
#define CT 32
__global__ __launch_bounds__(256) void scan_kernel(
    const float* __restrict__ u, const float* __restrict__ dt,
    const float* __restrict__ A_log, const float* __restrict__ xdbl,
    float* __restrict__ y)
{
    __shared__ float sBC[CT][2 * D_STATE];   // [t][0..63]=B, [t][64..127]=C

    const int tid = threadIdx.x;
    const int b   = blockIdx.x / (D_INNER / 256);
    const int d   = (blockIdx.x % (D_INNER / 256)) * 256 + tid;

    float Av[D_STATE], h[D_STATE];
    #pragma unroll
    for (int n = 0; n < D_STATE; ++n) {
        Av[n] = -__expf(A_log[(size_t)d * D_STATE + n]);
        h[n]  = 0.f;
    }

    for (int t0 = 0; t0 < SEQ; t0 += CT) {
        __syncthreads();
        // stage CT*128 floats = 1024 x 16B chunks, 4 per thread, async
        #pragma unroll
        for (int i = 0; i < 4; ++i) {
            int chunk = tid + i * 256;
            int tl = chunk >> 5;            // 32 chunks per timestep row
            int c4 = (chunk & 31) << 2;     // float offset within row
            async_b128(&sBC[tl][c4],
                       xdbl + (size_t)(b * SEQ + t0 + tl) * XDBL_W + DT_RANK + c4);
        }
        wait_async_0();
        __syncthreads();

        for (int tl = 0; tl < CT; ++tl) {
            size_t off = (size_t)(b * SEQ + t0 + tl) * D_INNER + d;
            float dtv = dt[off];
            float uv  = u[off];
            float du  = dtv * uv;
            float acc = 0.f;
            #pragma unroll
            for (int n = 0; n < D_STATE; ++n) {
                float dA = __expf(dtv * Av[n]);
                h[n] = dA * h[n] + du * sBC[tl][n];
                acc += h[n] * sBC[tl][D_STATE + n];
            }
            y[off] = acc;
        }
    }
}

// =====================================================================
// Gate: out16 = bf16( (y_scan + D[d]*x_conv) * silu(z) );  feeds out_proj only
// =====================================================================
__global__ __launch_bounds__(256) void gate_kernel(
    const float* __restrict__ yscan, const float* __restrict__ xc,
    const float* __restrict__ xz, const float* __restrict__ Dp,
    __bf16* __restrict__ out16)
{
    int idx = blockIdx.x * 256 + threadIdx.x;
    if (idx >= MROWS * D_INNER) return;
    int d   = idx % D_INNER;
    int row = idx / D_INNER;
    float z = xz[(size_t)row * (2 * D_INNER) + D_INNER + d];
    float v = (yscan[idx] + Dp[d] * xc[idx]) * (z / (1.f + __expf(-z)));
    out16[idx] = f2bf(v);
}

// =====================================================================
// Driver
// =====================================================================
extern "C" void kernel_launch(void* const* d_in, const int* in_sizes, int n_in,
                              void* d_out, int out_size, void* d_ws, size_t ws_size,
                              hipStream_t stream) {
    const float* x     = (const float*)d_in[0];
    const float* Wi    = (const float*)d_in[1];
    const float* cw    = (const float*)d_in[2];
    const float* cb    = (const float*)d_in[3];
    const float* Wx    = (const float*)d_in[4];
    const float* Wdt   = (const float*)d_in[5];
    const float* bdt   = (const float*)d_in[6];
    const float* A_log = (const float*)d_in[7];
    const float* Dp    = (const float*)d_in[8];
    const float* Wo    = (const float*)d_in[9];
    float* out = (float*)d_out;

    // ---- workspace carve-up ----
    char* ws = (char*)d_ws;
    size_t o = 0;
    auto alloc_f32 = [&](size_t n) { float* p = (float*)(ws + o); o += n * 4; return p; };
    auto alloc_b16 = [&](size_t n) { __bf16* p = (__bf16*)(ws + o); o += ((n * 2 + 15) & ~(size_t)15); return p; };

    float*  xz     = alloc_f32((size_t)MROWS * 2 * D_INNER);  // in_proj out
    float*  xc     = alloc_f32((size_t)MROWS * D_INNER);      // conv+silu (f32)
    float*  xdbl   = alloc_f32((size_t)MROWS * XDBL_W);       // x_proj out (f32)
    float*  dtb    = alloc_f32((size_t)MROWS * D_INNER);      // softplus'd dt
    float*  ysc    = alloc_f32((size_t)MROWS * D_INNER);      // scan out
    __bf16* xb16   = alloc_b16((size_t)MROWS * D_MODEL);      // layer-0 input bf16
    __bf16* xcb    = alloc_b16((size_t)MROWS * D_INNER);      // conv out bf16
    __bf16* xdbl16 = alloc_b16((size_t)MROWS * XDBL_W);       // x_proj out bf16
    __bf16* xbuf16 = alloc_b16((size_t)MROWS * D_MODEL);      // inter-layer act
    __bf16* wib    = alloc_b16((size_t)2 * D_INNER * D_MODEL);
    __bf16* wxb    = alloc_b16((size_t)XDBL_W * D_INNER);
    __bf16* wdtb   = alloc_b16((size_t)D_INNER * DT_RANK);
    __bf16* wob    = alloc_b16((size_t)D_MODEL * D_INNER);
    __bf16* ygb    = (__bf16*)dtb;   // dt is dead after scan; reuse for gate out

    dim3 blk(256);
    const int gM = (MROWS + BM - 1) / BM;   // 64
    auto cdiv = [](int a, int b) { return (a + b - 1) / b; };

    // initial input -> bf16 (once)
    cast_bf16_kernel<<<cdiv(MROWS * D_MODEL, 256), blk, 0, stream>>>(
        x, xb16, MROWS * D_MODEL);

    const __bf16* cur = xb16;
    for (int layer = 0; layer < DEPTH; ++layer) {
        const float* cw_l  = cw    + (size_t)layer * D_INNER * D_CONV;
        const float* cb_l  = cb    + (size_t)layer * D_INNER;
        const float* bdt_l = bdt   + (size_t)layer * D_INNER;
        const float* Al_l  = A_log + (size_t)layer * D_INNER * D_STATE;
        const float* Dp_l  = Dp    + (size_t)layer * D_INNER;

        // weights -> bf16 for this layer
        cast_bf16_kernel<<<cdiv(2 * D_INNER * D_MODEL, 256), blk, 0, stream>>>(
            Wi + (size_t)layer * 2 * D_INNER * D_MODEL, wib, 2 * D_INNER * D_MODEL);
        cast_bf16_kernel<<<cdiv(XDBL_W * D_INNER, 256), blk, 0, stream>>>(
            Wx + (size_t)layer * XDBL_W * D_INNER, wxb, XDBL_W * D_INNER);
        cast_bf16_kernel<<<cdiv(D_INNER * DT_RANK, 256), blk, 0, stream>>>(
            Wdt + (size_t)layer * D_INNER * DT_RANK, wdtb, D_INNER * DT_RANK);
        cast_bf16_kernel<<<cdiv(D_MODEL * D_INNER, 256), blk, 0, stream>>>(
            Wo + (size_t)layer * D_MODEL * D_INNER, wob, D_MODEL * D_INNER);

        // 1) in_proj: xz = cur @ Wi^T    [8192 x 3072, K=768]  (aligned)
        wmma_gemm_kernel<true><<<dim3((2 * D_INNER) / BN, gM), blk, 0, stream>>>(
            cur, D_MODEL, wib, D_MODEL, xz, nullptr, 2 * D_INNER,
            MROWS, 2 * D_INNER, D_MODEL, nullptr, 0);

        // 2) depthwise causal conv + SiLU
        conv_silu_kernel<<<(MROWS * D_INNER) / 256, blk, 0, stream>>>(
            xz, cw_l, cb_l, xc, xcb);

        // 3) x_proj: xdbl = xc @ Wx^T    [8192 x 176, K=1536]  (generic: N tail)
        wmma_gemm_kernel<false><<<dim3(cdiv(XDBL_W, BN), gM), blk, 0, stream>>>(
            xcb, D_INNER, wxb, D_INNER, xdbl, xdbl16, XDBL_W,
            MROWS, XDBL_W, D_INNER, nullptr, 0);

        // 4) dt = softplus(xdbl[:, :48] @ Wdt^T + bdt)  [K=48] (generic: K tail)
        wmma_gemm_kernel<false><<<dim3(D_INNER / BN, gM), blk, 0, stream>>>(
            xdbl16, XDBL_W, wdtb, DT_RANK, dtb, nullptr, D_INNER,
            MROWS, D_INNER, DT_RANK, bdt_l, 1);

        // 5) selective scan
        scan_kernel<<<BATCH * (D_INNER / 256), blk, 0, stream>>>(
            xc, dtb, Al_l, xdbl, ysc);

        // 6) gate -> bf16 (aliases dtb storage; dt is dead)
        gate_kernel<<<(MROWS * D_INNER) / 256, blk, 0, stream>>>(
            ysc, xc, xz, Dp_l, ygb);

        // 7) out_proj: dst = yg @ Wo^T   [8192 x 768, K=1536]  (aligned)
        bool last = (layer == DEPTH - 1);
        wmma_gemm_kernel<true><<<dim3(D_MODEL / BN, gM), blk, 0, stream>>>(
            ygb, D_INNER, wob, D_INNER,
            last ? out : nullptr, last ? nullptr : xbuf16, D_MODEL,
            MROWS, D_MODEL, D_INNER, nullptr, 0);

        cur = xbuf16;
    }
}
